// MultiHeadAttention_17918603559155
// MI455X (gfx1250) — matrix operand order repaired and checked by reading, not compiled, not run on hardware
//
#include <hip/hip_runtime.h>
#include <hip/hip_bf16.h>

// ---------------- problem constants ----------------
#define BB 4
#define SS 2048
#define DD 1024
#define HH 16
#define DKV 64
#define NEGV (-1000000000.0f)
#define LN_EPS 1e-6f

constexpr size_t W_ELEMS = (size_t)DD * DD;       // 1M
constexpr size_t T_ELEMS = (size_t)BB * SS * DD;  // 8M

typedef __attribute__((ext_vector_type(16))) _Float16 v16h;
typedef __attribute__((ext_vector_type(8)))  _Float16 v8h;
typedef __attribute__((ext_vector_type(8)))  float    v8f;
typedef __attribute__((ext_vector_type(4)))  int      v4i;

// ---------------- gfx1250 async LDS copy (ASYNCcnt path) ----------------
#if __has_builtin(__builtin_amdgcn_global_load_async_to_lds_b128) && \
    __has_builtin(__builtin_amdgcn_s_wait_asynccnt)
#define USE_ASYNC 1
typedef __attribute__((address_space(1))) v4i* gv4i_p;  // global <4 x i32>*
typedef __attribute__((address_space(3))) v4i* lv4i_p;  // LDS    <4 x i32>*
#else
#define USE_ASYNC 0
#endif

__device__ __forceinline__ void cp16_async(_Float16* l, const _Float16* g) {
#if USE_ASYNC
  __builtin_amdgcn_global_load_async_to_lds_b128((gv4i_p)(g), (lv4i_p)(l), 0,
                                                 0);
#else
  *(v8h*)l = *(const v8h*)g;
#endif
}

__device__ __forceinline__ void wait_async() {
#if USE_ASYNC
  __builtin_amdgcn_s_wait_asynccnt(0);
#endif
}

// ---------------- WMMA fragment helpers (k = 16h + s contiguous order for A and B) ----------
__device__ __forceinline__ v16h frag_a(const _Float16* p, int ld, int mBase,
                                       int kBase, int lane) {
  const int h = (lane >> 4) & 1, m = lane & 15;
  const _Float16* base = p + (mBase + m) * ld + kBase + h * 16;
  const v8h lo = *(const v8h*)(base);
  const v8h hi = *(const v8h*)(base + 8);
  return __builtin_shufflevector(lo, hi, 0, 1, 2, 3, 4, 5, 6, 7, 8, 9, 10, 11,
                                 12, 13, 14, 15);
}

__device__ __forceinline__ v16h frag_bt(const _Float16* p, int ld, int nBase,
                                        int kBase, int lane) {
  const int h = (lane >> 4) & 1, n = lane & 15;
  return *(const v16h*)(p + (nBase + n) * ld + kBase + h * 16);
}

__device__ __forceinline__ v8f wmma16(v16h a, v16h b, v8f c) {
  return __builtin_amdgcn_wmma_f32_16x16x32_f16(false, a, false, b, (short)0,
                                                c, false, false);
}

__device__ __forceinline__ v8h cvt8(const float* p) {
  const float4 a = *(const float4*)p;
  const float4 b = *(const float4*)(p + 4);
  v8h r;
  r[0] = (_Float16)a.x; r[1] = (_Float16)a.y;
  r[2] = (_Float16)a.z; r[3] = (_Float16)a.w;
  r[4] = (_Float16)b.x; r[5] = (_Float16)b.y;
  r[6] = (_Float16)b.z; r[7] = (_Float16)b.w;
  return r;
}

// ---------------- kernel 0a: weights fp32 -> f16, TRANSPOSED ----------------
__global__ __launch_bounds__(256) void convert_wT_kernel(
    const float* __restrict__ Wq, const float* __restrict__ Wk,
    const float* __restrict__ Wv, const float* __restrict__ Wo,
    _Float16* __restrict__ dstBase) {
  __shared__ float T[64][65];
  const int z = blockIdx.z;
  const float* src = (z == 0) ? Wq : (z == 1) ? Wk : (z == 2) ? Wv : Wo;
  _Float16* dst = dstBase + (size_t)z * W_ELEMS;
  const int rBase = blockIdx.y * 64, cBase = blockIdx.x * 64;
  const int tid = threadIdx.x;
  for (int i = tid; i < 64 * 64; i += 256) {
    const int r = i >> 6, c = i & 63;
    T[r][c] = src[(size_t)(rBase + r) * DD + cBase + c];
  }
  __syncthreads();
  for (int i = tid; i < 512; i += 256) {
    const int r = i >> 3, c0 = (i & 7) * 8;
    v8h out;
#pragma unroll
    for (int u = 0; u < 8; ++u) out[u] = (_Float16)T[c0 + u][r];
    *(v8h*)&dst[(size_t)(cBase + r) * DD + rBase + c0] = out;
  }
}

// ---------------- kernel 0b: activations fp32 -> f16 (same layout) ---------
__global__ void convert_act_kernel(const float* __restrict__ q,
                                   const float* __restrict__ k,
                                   const float* __restrict__ v,
                                   _Float16* __restrict__ dst) {
  const size_t tot8 = 3 * (T_ELEMS >> 3);
  for (size_t i = (size_t)blockIdx.x * blockDim.x + threadIdx.x; i < tot8;
       i += (size_t)gridDim.x * blockDim.x) {
    const size_t z = i >> 20;                       // T_ELEMS/8 == 1<<20
    const size_t off = (i & ((1u << 20) - 1)) << 3;
    const float* src = (z == 0) ? q : (z == 1) ? k : v;
    *(v8h*)&dst[z * T_ELEMS + off] = cvt8(&src[off]);
  }
}

// ---------------- kernel 1: fused QKV projections (double-buffered) --------
// z=0: qh=(q@Wq+bq)/8 -> (B,H,S,64); z=1: kh; z=2: vh TRANSPOSED (B,H,64,S).
__global__ __launch_bounds__(256) void qkv_proj_kernel(
    const _Float16* __restrict__ act16, const _Float16* __restrict__ WT,
    const float* __restrict__ bq, const float* __restrict__ bk,
    const float* __restrict__ bv, _Float16* __restrict__ qh,
    _Float16* __restrict__ kh, _Float16* __restrict__ vh) {
  __shared__ __align__(16) _Float16 As[2][64][40];
  __shared__ __align__(16) _Float16 Bt[2][64][40];

  const int tid = threadIdx.x, lane = tid & 31, w = tid >> 5;
  const int mt = w & 3, npair = w >> 2;
  const int rowBase = blockIdx.x * 64, colBase = blockIdx.y * 64;
  const int z = blockIdx.z;

  const _Float16* A = act16 + (size_t)z * T_ELEMS;
  const _Float16* WzT = WT + (size_t)z * W_ELEMS;
  const float* bias = (z == 0) ? bq : (z == 1) ? bk : bv;
  const float scale = (z == 0) ? 0.125f : 1.0f;

  const int ar = tid >> 2, ac = (tid & 3) * 8;  // 8 halves per thread

  auto stage = [&](int buf, int k0) {
    cp16_async(&As[buf][ar][ac], &A[(size_t)(rowBase + ar) * DD + k0 + ac]);
    cp16_async(&Bt[buf][ar][ac], &WzT[(size_t)(colBase + ar) * DD + k0 + ac]);
  };

  stage(0, 0);
  v8f c0 = {}, c1 = {};
  for (int t = 0; t < DD / 32; ++t) {
    wait_async();
    __syncthreads();
    if (t + 1 < DD / 32) stage((t + 1) & 1, (t + 1) * 32);
    const _Float16* Ab = &As[t & 1][0][0];
    const _Float16* Bb = &Bt[t & 1][0][0];
    const v16h a = frag_a(Ab, 40, mt * 16, 0, lane);
    c0 = wmma16(a, frag_bt(Bb, 40, (npair * 2 + 0) * 16, 0, lane), c0);
    c1 = wmma16(a, frag_bt(Bb, 40, (npair * 2 + 1) * 16, 0, lane), c1);
  }

  const int h = lane >> 4, n = lane & 15;
  const int col0 = colBase + (npair * 2 + 0) * 16 + n;
  const int col1 = colBase + (npair * 2 + 1) * 16 + n;

  if (z == 2) {
    const int sBase = (rowBase & (SS - 1)) + mt * 16 + 8 * h;
    const int bidx = rowBase >> 11;
    v8h v0, v1;
#pragma unroll
    for (int r = 0; r < 8; ++r) {
      v0[r] = (_Float16)(c0[r] + bias[col0]);
      v1[r] = (_Float16)(c1[r] + bias[col1]);
    }
    *(v8h*)&vh[((size_t)(bidx * HH + (col0 >> 6)) * DKV + (col0 & 63)) * SS +
               sBase] = v0;
    *(v8h*)&vh[((size_t)(bidx * HH + (col1 >> 6)) * DKV + (col1 & 63)) * SS +
               sBase] = v1;
  } else {
    _Float16* dst = (z == 0) ? qh : kh;
#pragma unroll
    for (int r = 0; r < 8; ++r) {
      const int grow = rowBase + mt * 16 + r + 8 * h;
      const int bidx = grow >> 11, s = grow & (SS - 1);
      dst[((size_t)(bidx * HH + (col0 >> 6)) * SS + s) * DKV + (col0 & 63)] =
          (_Float16)((c0[r] + bias[col0]) * scale);
      dst[((size_t)(bidx * HH + (col1 >> 6)) * SS + s) * DKV + (col1 & 63)] =
          (_Float16)((c1[r] + bias[col1]) * scale);
    }
  }
}

// ---------------- kernel 2: causal attention ----------------
__global__ __launch_bounds__(128) void attn_kernel(
    const _Float16* __restrict__ qh, const _Float16* __restrict__ kh,
    const _Float16* __restrict__ vh, _Float16* __restrict__ ctx,
    float* __restrict__ attn_out) {
  __shared__ __align__(16) _Float16 Qs[64][72];  // [q][dk]
  __shared__ __align__(16) _Float16 Ks[64][72];  // [key][dk]
  __shared__ __align__(16) _Float16 Vt[64][72];  // [dv][key]
  __shared__ __align__(16) _Float16 Ps[64][72];  // [q][key]

  const int tid = threadIdx.x, lane = tid & 31, w = tid >> 5;  // w in 0..3
  const int qb = blockIdx.x, bh = blockIdx.y;
  const int b = bh >> 4, hh = bh & 15;
  const int qBase = qb * 64;
  const int h = lane >> 4, n = lane & 15;

  const _Float16* Qp = qh + (size_t)bh * SS * DKV;
  const _Float16* Kp = kh + (size_t)bh * SS * DKV;
  const _Float16* VTp = vh + (size_t)bh * DKV * SS;  // [dv][s]

  const int sr = tid >> 1, sc = (tid & 1) * 32;  // 2 threads/row, 4 x 16B

#pragma unroll
  for (int u = 0; u < 4; ++u)
    cp16_async(&Qs[sr][sc + u * 8],
               &Qp[(size_t)(qBase + sr) * DKV + sc + u * 8]);
  wait_async();
  __syncthreads();
  const v16h aq0 = frag_a(&Qs[0][0], 72, w * 16, 0, lane);
  const v16h aq1 = frag_a(&Qs[0][0], 72, w * 16, 32, lane);

  // ---- pass 1: row max ----
  float pm[8];
#pragma unroll
  for (int r = 0; r < 8; ++r) pm[r] = -3.0e38f;

  for (int j = 0; j <= qb; ++j) {
#pragma unroll
    for (int u = 0; u < 4; ++u)
      cp16_async(&Ks[sr][sc + u * 8],
                 &Kp[(size_t)(j * 64 + sr) * DKV + sc + u * 8]);
    wait_async();
    __syncthreads();
#pragma unroll
    for (int nt = 0; nt < 4; ++nt) {
      v8f s = {};
      s = wmma16(aq0, frag_bt(&Ks[0][0], 72, nt * 16, 0, lane), s);
      s = wmma16(aq1, frag_bt(&Ks[0][0], 72, nt * 16, 32, lane), s);
#pragma unroll
      for (int r = 0; r < 8; ++r) {
        const int qrow = qBase + w * 16 + r + 8 * h;
        const int kcol = j * 64 + nt * 16 + n;
        pm[r] = fmaxf(pm[r], (kcol <= qrow) ? s[r] : NEGV);
      }
    }
    __syncthreads();
  }
#pragma unroll
  for (int r = 0; r < 8; ++r)
#pragma unroll
    for (int m = 8; m >= 1; m >>= 1)
      pm[r] = fmaxf(pm[r], __shfl_xor(pm[r], m, 16));

  // ---- pass 2: row sum + O accumulation ----
  float rs[8];
#pragma unroll
  for (int r = 0; r < 8; ++r) rs[r] = 0.0f;
  v8f O[4] = {};

  for (int j = 0; j <= qb; ++j) {
#pragma unroll
    for (int u = 0; u < 4; ++u) {
      cp16_async(&Ks[sr][sc + u * 8],
                 &Kp[(size_t)(j * 64 + sr) * DKV + sc + u * 8]);
      cp16_async(&Vt[sr][sc + u * 8],
                 &VTp[(size_t)sr * SS + j * 64 + sc + u * 8]);
    }
    wait_async();
    __syncthreads();
#pragma unroll
    for (int nt = 0; nt < 4; ++nt) {
      v8f s = {};
      s = wmma16(aq0, frag_bt(&Ks[0][0], 72, nt * 16, 0, lane), s);
      s = wmma16(aq1, frag_bt(&Ks[0][0], 72, nt * 16, 32, lane), s);
#pragma unroll
      for (int r = 0; r < 8; ++r) {
        const int qrow = qBase + w * 16 + r + 8 * h;
        const int kcol = j * 64 + nt * 16 + n;
        const float p = (kcol <= qrow) ? __expf(s[r] - pm[r]) : 0.0f;
        rs[r] += p;
        Ps[w * 16 + r + 8 * h][nt * 16 + n] = (_Float16)p;  // wave-local rows
      }
    }
#pragma unroll
    for (int kk = 0; kk < 2; ++kk) {
      const v16h ap = frag_a(&Ps[0][0], 72, w * 16, kk * 32, lane);
#pragma unroll
      for (int nt2 = 0; nt2 < 4; ++nt2)
        O[nt2] = wmma16(ap, frag_bt(&Vt[0][0], 72, nt2 * 16, kk * 32, lane),
                        O[nt2]);
    }
    __syncthreads();
  }
  float inv[8];
#pragma unroll
  for (int r = 0; r < 8; ++r) {
#pragma unroll
    for (int m = 8; m >= 1; m >>= 1) rs[r] += __shfl_xor(rs[r], m, 16);
    inv[r] = 1.0f / rs[r];
  }
#pragma unroll
  for (int nt2 = 0; nt2 < 4; ++nt2)
#pragma unroll
    for (int r = 0; r < 8; ++r) {
      const int qrow = qBase + w * 16 + r + 8 * h;
      ctx[((size_t)b * SS + qrow) * DD + hh * DKV + nt2 * 16 + n] =
          (_Float16)(O[nt2][r] * inv[r]);
    }

  // ---- pass 3: write normalized attn (fp32) ----
  for (int j = 0; j <= qb; ++j) {
#pragma unroll
    for (int u = 0; u < 4; ++u)
      cp16_async(&Ks[sr][sc + u * 8],
                 &Kp[(size_t)(j * 64 + sr) * DKV + sc + u * 8]);
    wait_async();
    __syncthreads();
#pragma unroll
    for (int nt = 0; nt < 4; ++nt) {
      v8f s = {};
      s = wmma16(aq0, frag_bt(&Ks[0][0], 72, nt * 16, 0, lane), s);
      s = wmma16(aq1, frag_bt(&Ks[0][0], 72, nt * 16, 32, lane), s);
#pragma unroll
      for (int r = 0; r < 8; ++r) {
        const int qrow = qBase + w * 16 + r + 8 * h;
        const int kcol = j * 64 + nt * 16 + n;
        const float p =
            (kcol <= qrow) ? __expf(s[r] - pm[r]) * inv[r] : 0.0f;
        attn_out[(size_t)(bh * SS + qrow) * SS + kcol] = p;
      }
    }
    __syncthreads();
  }

  // zero-fill fully masked columns (float4 stores, no runtime div)
  const int kstart = (qb + 1) * 64;
  if (kstart < SS) {
    const int r = tid >> 1;  // 2 threads per row
    const float4 z4 = make_float4(0.f, 0.f, 0.f, 0.f);
    float* rowp = attn_out + (size_t)(bh * SS + qBase + r) * SS;
    for (int c = kstart + (tid & 1) * 4; c < SS; c += 8)
      *reinterpret_cast<float4*>(rowp + c) = z4;
  }
}

// ---------------- kernel 3: output projection + residual (double-buffered) -
__global__ __launch_bounds__(256) void out_proj_kernel(
    const _Float16* __restrict__ ctx, const _Float16* __restrict__ WoT,
    const float* __restrict__ bo, const float* __restrict__ resid,
    float* __restrict__ xbuf) {
  __shared__ __align__(16) _Float16 As[2][64][40];
  __shared__ __align__(16) _Float16 Bt[2][64][40];

  const int tid = threadIdx.x, lane = tid & 31, w = tid >> 5;
  const int mt = w & 3, npair = w >> 2;
  const int rowBase = blockIdx.x * 64, colBase = blockIdx.y * 64;
  const int ar = tid >> 2, ac = (tid & 3) * 8;

  auto stage = [&](int buf, int k0) {
    cp16_async(&As[buf][ar][ac], &ctx[(size_t)(rowBase + ar) * DD + k0 + ac]);
    cp16_async(&Bt[buf][ar][ac], &WoT[(size_t)(colBase + ar) * DD + k0 + ac]);
  };

  stage(0, 0);
  v8f c0 = {}, c1 = {};
  for (int t = 0; t < DD / 32; ++t) {
    wait_async();
    __syncthreads();
    if (t + 1 < DD / 32) stage((t + 1) & 1, (t + 1) * 32);
    const _Float16* Ab = &As[t & 1][0][0];
    const _Float16* Bb = &Bt[t & 1][0][0];
    const v16h a = frag_a(Ab, 40, mt * 16, 0, lane);
    c0 = wmma16(a, frag_bt(Bb, 40, (npair * 2 + 0) * 16, 0, lane), c0);
    c1 = wmma16(a, frag_bt(Bb, 40, (npair * 2 + 1) * 16, 0, lane), c1);
  }

  const int h = lane >> 4, n = lane & 15;
  const int col0 = colBase + (npair * 2 + 0) * 16 + n;
  const int col1 = colBase + (npair * 2 + 1) * 16 + n;
#pragma unroll
  for (int r = 0; r < 8; ++r) {
    const int grow = rowBase + mt * 16 + r + 8 * h;
    xbuf[(size_t)grow * DD + col0] =
        c0[r] + bo[col0] + resid[(size_t)grow * DD + col0];
    xbuf[(size_t)grow * DD + col1] =
        c1[r] + bo[col1] + resid[(size_t)grow * DD + col1];
  }
}

// ---------------- kernel 4: layernorm ----------------
__global__ __launch_bounds__(256) void layernorm_kernel(
    const float* __restrict__ xbuf, const float* __restrict__ gamma,
    const float* __restrict__ beta, float* __restrict__ y) {
  __shared__ float redS[8], redQ[8];
  const int row = blockIdx.x, tid = threadIdx.x;
  const float* xr = xbuf + (size_t)row * DD;

  float xv[4], sum = 0.f, sq = 0.f;
#pragma unroll
  for (int kk = 0; kk < 4; ++kk) {
    const float x = xr[tid + kk * 256];
    xv[kk] = x;
    sum += x;
    sq += x * x;
  }
#pragma unroll
  for (int m = 16; m >= 1; m >>= 1) {
    sum += __shfl_xor(sum, m, 32);
    sq += __shfl_xor(sq, m, 32);
  }
  if ((tid & 31) == 0) {
    redS[tid >> 5] = sum;
    redQ[tid >> 5] = sq;
  }
  __syncthreads();
  float ts = 0.f, tq = 0.f;
#pragma unroll
  for (int i = 0; i < 8; ++i) {
    ts += redS[i];
    tq += redQ[i];
  }
  const float mu = ts * (1.0f / DD);
  const float var = tq * (1.0f / DD) - mu * mu;
  const float rstd = rsqrtf(var + LN_EPS);
#pragma unroll
  for (int kk = 0; kk < 4; ++kk) {
    const int c = tid + kk * 256;
    y[(size_t)row * DD + c] = (xv[kk] - mu) * rstd * gamma[c] + beta[c];
  }
}

// ---------------- host launch ----------------
extern "C" void kernel_launch(void* const* d_in, const int* in_sizes, int n_in,
                              void* d_out, int out_size, void* d_ws,
                              size_t ws_size, hipStream_t stream) {
  const float* q = (const float*)d_in[0];
  const float* k = (const float*)d_in[1];
  const float* v = (const float*)d_in[2];
  // d_in[3] = mask (causal; implemented analytically)
  const float* Wq = (const float*)d_in[4];
  const float* bq = (const float*)d_in[5];
  const float* Wk = (const float*)d_in[6];
  const float* bk = (const float*)d_in[7];
  const float* Wv = (const float*)d_in[8];
  const float* bv = (const float*)d_in[9];
  const float* Wo = (const float*)d_in[10];
  const float* bo = (const float*)d_in[11];
  const float* gamma = (const float*)d_in[12];
  const float* beta = (const float*)d_in[13];

  float* y_out = (float*)d_out;
  float* attn_out = y_out + T_ELEMS;

  _Float16* WT = (_Float16*)d_ws;      // 4 transposed f16 weights
  _Float16* act16 = WT + 4 * W_ELEMS;  // q,k,v in f16
  _Float16* qh = act16 + 3 * T_ELEMS;
  _Float16* kh = qh + T_ELEMS;
  _Float16* vh = kh + T_ELEMS;  // transposed (B,H,64,S)
  _Float16* ctx = vh + T_ELEMS;
  float* xbuf = (float*)(ctx + T_ELEMS);

  convert_wT_kernel<<<dim3(16, 16, 4), 256, 0, stream>>>(Wq, Wk, Wv, Wo, WT);
  convert_act_kernel<<<3072, 256, 0, stream>>>(q, k, v, act16);

  qkv_proj_kernel<<<dim3(BB * SS / 64, DD / 64, 3), 256, 0, stream>>>(
      act16, WT, bq, bk, bv, qh, kh, vh);

  attn_kernel<<<dim3(SS / 64, BB * HH), 128, 0, stream>>>(qh, kh, vh, ctx,
                                                          attn_out);

  out_proj_kernel<<<dim3(BB * SS / 64, DD / 64), 256, 0, stream>>>(
      ctx, WT + 3 * W_ELEMS, bo, q, xbuf);

  layernorm_kernel<<<BB * SS, 256, 0, stream>>>(xbuf, gamma, beta, y_out);
}
